// STHGNN_62577673503470
// MI455X (gfx1250) — compile-verified, matrix operand order checked by reading, not run on hardware
//
#include <hip/hip_runtime.h>
#include <hip/hip_bf16.h>
#include <math.h>

#define B_ 16
#define L_ 64
#define N_ 512
#define F_ 64
#define D_ 64
#define E_ 40
#define TOPK_ 10
#define NSLICE (B_*L_)
#define ALPHA_ 3.0f
#define NEG_ 0.2f

typedef __bf16 bf16;
typedef __attribute__((ext_vector_type(16))) __bf16 v16bf;
typedef __attribute__((ext_vector_type(8)))  __bf16 v8bf;
typedef __attribute__((ext_vector_type(8)))  float  v8f;
typedef __attribute__((ext_vector_type(4)))  float  v4f;

__device__ __forceinline__ bf16  tobf(float f) { return (bf16)f; }
__device__ __forceinline__ float fbf(bf16 b)   { return (float)b; }

__device__ __forceinline__ v8f wmma_bf16(v16bf a, v16bf b, v8f c) {
    // D(16x16,f32) = A(16x32,bf16) x B(32x16,bf16) + C
    return __builtin_amdgcn_wmma_f32_16x16x32_bf16(false, a, false, b, (short)0, c, false, false);
}

// bf16 fragment from a row-major bf16 matrix (pre-transposed weights).
// ISA 16-bit A layout: lane<16 -> row nl, K chunks {kb..kb+7, kb+16..kb+23};
// lane>=16 -> same row, chunks shifted by 8. B mirrored on Wt[n][k] = W[k][n].
__device__ __forceinline__ v16bf frag_rm(const bf16* m, int rowbase, int str, int kb,
                                         int nl, int hi) {
    const bf16* p = m + (rowbase + nl) * str + kb + hi * 8;
    v8bf lo = *(const v8bf*)(p);
    v8bf hh = *(const v8bf*)(p + 16);
    v16bf r;
#pragma unroll
    for (int j = 0; j < 8; ++j) { r[j] = lo[j]; r[j + 8] = hh[j]; }
    return r;
}

// bf16 fragment from a row-major f32 matrix in LDS (convert at fragment build).
__device__ __forceinline__ v16bf frag_f32(const float* m, int rowbase, int str, int kb,
                                          int nl, int hi) {
    const float* p = m + (rowbase + nl) * str + kb + hi * 8;
    v8f lo = *(const v8f*)(p);
    v8f hh = *(const v8f*)(p + 16);
    v16bf r;
#pragma unroll
    for (int j = 0; j < 8; ++j) { r[j] = tobf(lo[j]); r[j + 8] = tobf(hh[j]); }
    return r;
}

// Gather x_neigh row (CSR over hyperedges) directly into the two K-step A fragments.
__device__ __forceinline__ void gather_np(const bf16* she, const int* offs, const int* nlist,
                                          const float* invdeg,
                                          int tb, int nl, int hi, v16bf& f0, v16bf& f1) {
    int r = tb + nl;
    int s = offs[r], e = offs[r + 1];
    float inv = invdeg[r];
    float acc[32];
#pragma unroll
    for (int j = 0; j < 32; ++j) acc[j] = 0.0f;
    int o = hi * 8;
    for (int t = s; t < e; ++t) {
        int m = nlist[t];
        const bf16* row = she + m * F_ + o;
#pragma unroll
        for (int c = 0; c < 4; ++c) {
            v8bf v = *(const v8bf*)(row + c * 16);
#pragma unroll
            for (int j = 0; j < 8; ++j) acc[c * 8 + j] += fbf(v[j]);
        }
    }
#pragma unroll
    for (int j = 0; j < 8; ++j) {
        f0[j]     = tobf(acc[j]      * inv);
        f0[j + 8] = tobf(acc[8 + j]  * inv);
        f1[j]     = tobf(acc[16 + j] * inv);
        f1[j + 8] = tobf(acc[24 + j] * inv);
    }
}

// ---------------- setup kernels (tiny; topology shared by all 1024 slices) -------------

__global__ void k_zero(int* ndeg) {
    int g = blockIdx.x * 256 + threadIdx.x;
    if (g < N_) ndeg[g] = 0;
}

__global__ void k_feats(const int* __restrict__ idx, const float* __restrict__ emb1,
                        const float* __restrict__ w, const float* __restrict__ b,
                        float* __restrict__ feats) {
    int g = blockIdx.x * 256 + threadIdx.x;
    if (g >= N_ * E_) return;
    int n = g / E_, e = g % E_;
    int in = idx[n];
    float acc = b[e];
    for (int k = 0; k < E_; ++k) acc += emb1[in * E_ + k] * w[k * E_ + e];
    feats[g] = tanhf(ALPHA_ * acc);
}

__global__ __launch_bounds__(256) void k_topk(const float* __restrict__ feats,
                                              int* __restrict__ topidx,
                                              int* __restrict__ ndeg) {
    __shared__ float sv[N_];
    __shared__ float rv[256];
    __shared__ int   ri[256];
    int m = blockIdx.x, tid = threadIdx.x;
    for (int c = tid; c < N_; c += 256) {
        float s = 0.0f;
        for (int k = 0; k < E_; ++k) s += feats[m * E_ + k] * feats[c * E_ + k];
        sv[c] = s;
    }
    __syncthreads();
    for (int pass = 0; pass < TOPK_; ++pass) {
        float b0 = sv[tid]; int i0 = tid;
        float b1 = sv[tid + 256];
        if (b1 > b0) { b0 = b1; i0 = tid + 256; }   // ties -> lower index (matches top_k)
        rv[tid] = b0; ri[tid] = i0;
        __syncthreads();
        for (int s = 128; s > 0; s >>= 1) {
            if (tid < s) {
                if (rv[tid + s] > rv[tid] ||
                    (rv[tid + s] == rv[tid] && ri[tid + s] < ri[tid])) {
                    rv[tid] = rv[tid + s]; ri[tid] = ri[tid + s];
                }
            }
            __syncthreads();
        }
        if (tid == 0) {
            int wdx = ri[0];
            topidx[m * TOPK_ + pass] = wdx;
            sv[wdx] = -3.4e38f;
            atomicAdd(&ndeg[wdx], 1);
        }
        __syncthreads();
    }
}

__global__ __launch_bounds__(512) void k_scan(const int* __restrict__ ndeg,
                                              int* __restrict__ offs,
                                              float* __restrict__ invdeg) {
    __shared__ int t[N_];
    int i = threadIdx.x;
    t[i] = ndeg[i];
    __syncthreads();
    for (int o = 1; o < N_; o <<= 1) {
        int v = (i >= o) ? t[i - o] : 0;
        __syncthreads();
        t[i] += v;
        __syncthreads();
    }
    offs[i + 1] = t[i];
    if (i == 0) offs[0] = 0;
    int d = ndeg[i]; if (d < 1) d = 1;
    invdeg[i] = 1.0f / (float)d;          // precomputed once; removes div from hot loop
}

// Deterministic CSR fill (no atomics -> bitwise-reproducible summation order).
__global__ void k_fill(const int* __restrict__ topidx, const int* __restrict__ offs,
                       int* __restrict__ nlist) {
    int n = blockIdx.x * 256 + threadIdx.x;
    if (n >= N_) return;
    int pos = offs[n];
    for (int e = 0; e < N_; ++e) {
#pragma unroll
        for (int k = 0; k < TOPK_; ++k) {
            if (topidx[e * TOPK_ + k] == n) nlist[pos++] = e;
        }
    }
}

// Transposed bf16 weights so WMMA B fragments are contiguous 16B loads.
__global__ void k_wtrans(const float* __restrict__ Ws, const float* __restrict__ Wn,
                         const float* __restrict__ Wc,
                         bf16* __restrict__ WtS, bf16* __restrict__ WtN,
                         bf16* __restrict__ WtC) {
    int g = blockIdx.x * 256 + threadIdx.x;
    if (g < F_ * D_) {
        int n = g / F_, k = g % F_;
        WtS[n * F_ + k] = tobf(Ws[k * D_ + n]);
        WtN[n * F_ + k] = tobf(Wn[k * D_ + n]);
    }
    if (g < D_ * 2 * D_) {
        int n = g / (2 * D_), kk = g % (2 * D_);
        WtC[n * (2 * D_) + kk] = tobf(Wc[kk * D_ + n]);
    }
}

// ---------------- main fused kernel: one (b,l) slice per workgroup ---------------------

__global__ __launch_bounds__(256) void k_main(const float* __restrict__ x,
                                              const float* __restrict__ att_g,
                                              const int* __restrict__ topidx,
                                              const int* __restrict__ offs,
                                              const int* __restrict__ nlist,
                                              const float* __restrict__ invdeg,
                                              const bf16* __restrict__ WtS,
                                              const bf16* __restrict__ WtN,
                                              const bf16* __restrict__ WtC,
                                              float* __restrict__ out) {
    __shared__ float sxf[N_ * F_];     // x slice, f32 (128 KB) — TDM destination
    __shared__ bf16  she[N_ * F_];     // hyperedge features, bf16 (64 KB)
    __shared__ float satt[2 * D_];
    __shared__ float sscore[N_];
    __shared__ float sred[256];
    __shared__ float s_stat[1];

    const int tid  = threadIdx.x;
    const int lane = tid & 31;
    const int wv   = tid >> 5;
    const int nl   = lane & 15;
    const int hi   = lane >> 4;
    const int slice = blockIdx.x;
    const float* gx = x + (size_t)slice * (N_ * F_);

    for (int i = tid; i < 2 * D_; i += 256) satt[i] = att_g[i];

    // stage 0: stage the whole 128 KB x slice into LDS.
#if __has_builtin(__builtin_amdgcn_tensor_load_to_lds)
    // Tensor Data Mover: one DMA descriptor per workgroup (ISA 08_async_tensor §8).
    // D# group0: count=1 | lds_addr | global_addr | type=2.  group1: 1-D tensor,
    // data_size=4B, tensor_dim0 = tile_dim0 = 32768 elems, dim1=1, stride=32768.
    if (wv == 0) {
        typedef unsigned int u32x4 __attribute__((ext_vector_type(4)));
        typedef int          i32x8 __attribute__((ext_vector_type(8)));
        typedef int          i32x4 __attribute__((ext_vector_type(4)));
        unsigned long long ga = (unsigned long long)(uintptr_t)gx;
        unsigned int laddr = (unsigned int)(uintptr_t)&sxf[0];  // low 32b = LDS byte offset
        u32x4 g0 = { 1u, laddr, (unsigned int)ga,
                     (unsigned int)(ga >> 32) | 0x80000000u };  // type=2 in bits 127:126
        i32x8 g1 = { 0x20000,            // data_size=2 (4B), mask=0, no pad/iterate
                     (int)0x80000000u,   // tensor_dim0[15:0]=0x8000 in bits 79:48 (low half)
                     0x00010000,         // tensor_dim1 = 1 (bits 111:80)
                     (int)0x80000000u,   // tile_dim0 = 0x8000 (bits 127:112)
                     0,                  // tile_dim1/2 = 0 (unused)
                     32768,              // tensor_dim0_stride[31:0]
                     0, 0 };
        i32x4 gz4 = { 0, 0, 0, 0 };
        i32x8 gz8 = { 0, 0, 0, 0, 0, 0, 0, 0 };
        __builtin_amdgcn_tensor_load_to_lds(g0, g1, gz4, gz4, gz8, 0);
        __builtin_amdgcn_s_wait_tensorcnt(0);
    }
#else
    for (int i = tid * 4; i < N_ * F_; i += 256 * 4) {
        *(v4f*)&sxf[i] = *(const v4f*)(gx + i);
    }
#endif
    __syncthreads();

    // stage 1: hyperedge features: mean of the 10 member nodes, LeakyReLU(0.2)
    for (int m = wv; m < N_; m += 8) {
        const int* mem = topidx + m * TOPK_;
        float a0 = 0.0f, a1 = 0.0f;
#pragma unroll
        for (int k = 0; k < TOPK_; ++k) {
            int id = mem[k];
            a0 += sxf[id * F_ + lane];
            a1 += sxf[id * F_ + lane + 32];
        }
        a0 *= 0.1f; a1 *= 0.1f;                 // e_deg == 10 exactly
        a0 = (a0 >= 0.0f) ? a0 : NEG_ * a0;
        a1 = (a1 >= 0.0f) ? a1 : NEG_ * a1;
        she[m * F_ + lane]      = tobf(a0);
        she[m * F_ + lane + 32] = tobf(a1);
    }
    __syncthreads();

    // stage 2: per 16-row tile: WMMA projections + fused attention scores
    for (int mt = wv; mt < 32; mt += 8) {
        int tb = mt * 16;
        v16bf as0 = frag_f32(sxf, tb, F_, 0,  nl, hi);
        v16bf as1 = frag_f32(sxf, tb, F_, 32, nl, hi);
        v16bf an0, an1;
        gather_np(she, offs, nlist, invdeg, tb, nl, hi, an0, an1);
        float sc[8];
#pragma unroll
        for (int v = 0; v < 8; ++v) sc[v] = 0.0f;
#pragma unroll
        for (int nt = 0; nt < 4; ++nt) {
            v8f cs = {}; v8f cn = {};
            cs = wmma_bf16(as0, frag_rm(WtS, nt * 16, F_, 0,  nl, hi), cs);
            cs = wmma_bf16(as1, frag_rm(WtS, nt * 16, F_, 32, nl, hi), cs);
            cn = wmma_bf16(an0, frag_rm(WtN, nt * 16, F_, 0,  nl, hi), cn);
            cn = wmma_bf16(an1, frag_rm(WtN, nt * 16, F_, 32, nl, hi), cn);
            int col = nt * 16 + nl;
            float aS = satt[col], aN = satt[D_ + col];
#pragma unroll
            for (int v = 0; v < 8; ++v) {
                float p = cs[v] * aS + cn[v] * aN;      // score partial for row v+8*hi
                p += __shfl_xor(p, 1);
                p += __shfl_xor(p, 2);
                p += __shfl_xor(p, 4);
                p += __shfl_xor(p, 8);                  // reduce within 16-lane group
                sc[v] += p;
            }
        }
        if (nl == 0) {
#pragma unroll
            for (int v = 0; v < 8; ++v) sscore[tb + v + 8 * hi] = sc[v];
        }
    }
    __syncthreads();

    // stage 3: softmax over the 512 nodes
    {
        float m0 = fmaxf(sscore[tid], sscore[tid + 256]);
        sred[tid] = m0;
        __syncthreads();
        for (int s = 128; s > 0; s >>= 1) {
            if (tid < s) sred[tid] = fmaxf(sred[tid], sred[tid + s]);
            __syncthreads();
        }
        float mx = sred[0];
        __syncthreads();
        float e0 = __expf(sscore[tid] - mx);
        float e1 = __expf(sscore[tid + 256] - mx);
        sscore[tid] = e0; sscore[tid + 256] = e1;
        sred[tid] = e0 + e1;
        __syncthreads();
        for (int s = 128; s > 0; s >>= 1) {
            if (tid < s) sred[tid] += sred[tid + s];
            __syncthreads();
        }
        if (tid == 0) s_stat[0] = 1.0f / sred[0];
        __syncthreads();
    }
    float invs = s_stat[0];

    // stage 4: out = diag(softmax) * (x_cat @ W_concat): 4 accumulating WMMAs (K=128)
    for (int mt = wv; mt < 32; mt += 8) {
        int tb = mt * 16;
        v16bf as0 = frag_f32(sxf, tb, F_, 0,  nl, hi);
        v16bf as1 = frag_f32(sxf, tb, F_, 32, nl, hi);
        v16bf an0, an1;
        gather_np(she, offs, nlist, invdeg, tb, nl, hi, an0, an1);
        float wr[8];
#pragma unroll
        for (int v = 0; v < 8; ++v) wr[v] = sscore[tb + v + 8 * hi] * invs;
        float* og = out + (size_t)slice * (N_ * D_) + (size_t)tb * D_;
#pragma unroll
        for (int nt = 0; nt < 4; ++nt) {
            v8f cy = {};
            cy = wmma_bf16(as0, frag_rm(WtC, nt * 16, 2 * D_, 0,  nl, hi), cy);
            cy = wmma_bf16(as1, frag_rm(WtC, nt * 16, 2 * D_, 32, nl, hi), cy);
            cy = wmma_bf16(an0, frag_rm(WtC, nt * 16, 2 * D_, 64, nl, hi), cy);
            cy = wmma_bf16(an1, frag_rm(WtC, nt * 16, 2 * D_, 96, nl, hi), cy);
            int col = nt * 16 + nl;
#pragma unroll
            for (int v = 0; v < 8; ++v) {
                int row = v + 8 * hi;
                og[row * D_ + col] = cy[v] * wr[v];
            }
        }
    }
}

// ---------------- launch ---------------------------------------------------------------

extern "C" void kernel_launch(void* const* d_in, const int* in_sizes, int n_in,
                              void* d_out, int out_size, void* d_ws, size_t ws_size,
                              hipStream_t stream) {
    const float* x       = (const float*)d_in[0];
    const int*   idx     = (const int*)  d_in[1];
    const float* emb1    = (const float*)d_in[2];
    const float* lin1_w  = (const float*)d_in[3];
    const float* lin1_b  = (const float*)d_in[4];
    const float* W_self  = (const float*)d_in[5];
    const float* W_neigh = (const float*)d_in[6];
    const float* W_cat   = (const float*)d_in[7];
    const float* att     = (const float*)d_in[8];
    float* out = (float*)d_out;
    char*  ws  = (char*)d_ws;

    float* feats  = (float*)(ws + 0);        // 512*40 f32
    int*   topidx = (int*)  (ws + 81920);    // 512*10
    int*   ndeg   = (int*)  (ws + 102400);   // 512
    int*   offs   = (int*)  (ws + 104448);   // 513
    int*   nlist  = (int*)  (ws + 106512);   // 5120
    bf16*  WtS    = (bf16*) (ws + 126992);   // 64x64
    bf16*  WtN    = (bf16*) (ws + 135184);   // 64x64
    bf16*  WtC    = (bf16*) (ws + 143376);   // 64x128
    float* invdeg = (float*)(ws + 159760);   // 512 f32

    k_zero  <<<2, 256, 0, stream>>>(ndeg);
    k_feats <<<(N_ * E_ + 255) / 256, 256, 0, stream>>>(idx, emb1, lin1_w, lin1_b, feats);
    k_topk  <<<N_, 256, 0, stream>>>(feats, topidx, ndeg);
    k_scan  <<<1, 512, 0, stream>>>(ndeg, offs, invdeg);
    k_fill  <<<2, 256, 0, stream>>>(topidx, offs, nlist);
    k_wtrans<<<32, 256, 0, stream>>>(W_self, W_neigh, W_cat, WtS, WtN, WtC);
    k_main  <<<NSLICE, 256, 0, stream>>>(x, att, topidx, offs, nlist, invdeg,
                                         WtS, WtN, WtC, out);
}
